// StructureTransformer_28733331210485
// MI455X (gfx1250) — matrix operand order repaired and checked by reading, not compiled, not run on hardware
//
#include <hip/hip_runtime.h>
#include <hip/hip_bf16.h>
#include <math.h>

typedef __bf16 bf16_t;
typedef __attribute__((ext_vector_type(16))) __bf16 v16bf;
typedef __attribute__((ext_vector_type(8)))  float  v8f;

#define B_   2
#define N_   512
#define CS_  384
#define CZ_  128
#define H_   12
#define CH_  32
#define PQ_  4
#define PV_  8
#define HID_ 1536
#define BN_  (B_*N_)
#define BH_  (B_*H_)

__device__ __forceinline__ float  bf2f(bf16_t x){ return (float)x; }
__device__ __forceinline__ bf16_t f2bf(float  x){ return (bf16_t)x; }

// ---------------------------------------------------------------------------
// Generic batched bf16 WMMA GEMM:  C[z] = alpha(z) * A[z] @ Bw[z]^T + beta*C[z]
// A: (M x K) row-major bf16.  Bw: (N x K) row-major bf16 (i.e. weights (out,in)).
// Requires M%64==0, N%64==0, K%32==0 (all call sites satisfy this).
// Block: 256 threads = 8 waves; 64x64 C tile; each wave owns a 16x32 sub-tile
// (two v_wmma_f32_16x16x32_bf16 accumulators).
// ---------------------------------------------------------------------------
__global__ __launch_bounds__(256) void k_gemm_bf16(
    const bf16_t* __restrict__ A, const bf16_t* __restrict__ Bw,
    float* __restrict__ C, int M, int N, int K,
    long long sA, long long sB, long long sC,
    float alphaS, const float* __restrict__ alphaPtr, float beta)
{
  const int z = blockIdx.z;
  A  += (long long)z * sA;
  Bw += (long long)z * sB;
  C  += (long long)z * sC;
  const int tile_n = blockIdx.x * 64;
  const int tile_m = blockIdx.y * 64;

  __shared__ __align__(16) bf16_t As[64][48];
  __shared__ __align__(16) bf16_t Bs[64][48];

  const int tid  = threadIdx.x;
  const int lane = tid & 31;
  const int wave = tid >> 5;
  const int wm   = wave & 3;   // M sub-tile (16 rows)
  const int wn   = wave >> 2;  // N sub-tile (32 cols)

  v8f acc0 = {0.f,0.f,0.f,0.f,0.f,0.f,0.f,0.f};
  v8f acc1 = {0.f,0.f,0.f,0.f,0.f,0.f,0.f,0.f};

  const int lrow = tid >> 2;        // 0..63
  const int lk   = (tid & 3) * 8;   // 0,8,16,24 (8 bf16 = 16 bytes)

  for (int k0 = 0; k0 < K; k0 += 32) {
    const float4* ga = (const float4*)(A  + (long long)(tile_m + lrow) * K + k0 + lk);
    const float4* gb = (const float4*)(Bw + (long long)(tile_n + lrow) * K + k0 + lk);
    *(float4*)(&As[lrow][lk]) = *ga;
    *(float4*)(&Bs[lrow][lk]) = *gb;
    if (k0 + 32 < K) {
      __builtin_prefetch((const void*)(A  + (long long)(tile_m + lrow) * K + k0 + 32 + lk), 0, 1);
      __builtin_prefetch((const void*)(Bw + (long long)(tile_n + lrow) * K + k0 + 32 + lk), 0, 1);
    }
    __syncthreads();

    // A fragment (16x32 bf16): lanes 0-15 hold M=0..15 K{0..7,16..23};
    // lanes 16-31 hold K{8..15,24..31}.
    v16bf af;
    {
      const int m = wm * 16 + (lane & 15);
      const int khalf = (lane < 16) ? 0 : 8;
#pragma unroll
      for (int j = 0; j < 8; ++j) {
        const int kk = (j < 4) ? (khalf + 2*j) : (16 + khalf + 2*(j-4));
        af[2*j]   = As[m][kk];
        af[2*j+1] = As[m][kk+1];
      }
    }
    const int kb0 = (lane < 16) ? 0 : 8;
#pragma unroll
    for (int ns = 0; ns < 2; ++ns) {
      const int n = wn * 32 + ns * 16 + (lane & 15);
      v16bf bfr;
#pragma unroll
      for (int j = 0; j < 8; ++j) {
        bfr[2*j]   = Bs[n][kb0 + 2*j + ((j < 4) ? 0 : 8)];
        bfr[2*j+1] = Bs[n][kb0 + 2*j + ((j < 4) ? 0 : 8) + 1];
      }
      if (ns == 0)
        acc0 = __builtin_amdgcn_wmma_f32_16x16x32_bf16(false, af, false, bfr,
                                                       (short)0, acc0, false, false);
      else
        acc1 = __builtin_amdgcn_wmma_f32_16x16x32_bf16(false, af, false, bfr,
                                                       (short)0, acc1, false, false);
    }
    __syncthreads();
  }

  const float alpha = alphaS * (alphaPtr ? alphaPtr[z] : 1.0f);
  // C/D f32 layout: VGPR r -> (M=r, N=lane) lanes 0-15 ; (M=r+8, N=lane-16) lanes 16-31
  const int col0 = tile_n + wn * 32 + (lane & 15);
  const int row0 = tile_m + wm * 16 + ((lane < 16) ? 0 : 8);
#pragma unroll
  for (int r = 0; r < 8; ++r) {
    long long i0 = (long long)(row0 + r) * N + col0;
    long long i1 = i0 + 16;
    float v0 = alpha * acc0[r];
    float v1 = alpha * acc1[r];
    if (beta != 0.f) { v0 += beta * C[i0]; v1 += beta * C[i1]; }
    C[i0] = v0; C[i1] = v1;
  }
}

// ---------------------------------------------------------------------------
__global__ void k_f32_to_bf16(const float* __restrict__ src, bf16_t* __restrict__ dst, int n){
  int i = blockIdx.x * blockDim.x + threadIdx.x;
  if (i < n) dst[i] = f2bf(src[i]);
}

__global__ void k_copy_f32(const float* __restrict__ src, float* __restrict__ dst, int n){
  int i = blockIdx.x * blockDim.x + threadIdx.x;
  if (i < n) dst[i] = src[i];
}

// gammaArr[z] = wL * wC * softplus(head_w[z % H]) for z in [0, B*H)
__global__ void k_gamma(const float* __restrict__ head_w, float* __restrict__ gammaArr){
  int i = threadIdx.x;
  if (i < BH_) {
    float hw = head_w[i % H_];
    float sp = logf(1.f + expf(hw));
    gammaArr[i] = 0.57735026919f /*wL*/ * 0.23570226039f /*wC=sqrt(1/18)*/ * sp;
  }
}

// out_bf16[row] = rms_norm(x[row], w) ; one block per row
__global__ __launch_bounds__(256) void k_rmsnorm_bf16(
    const float* __restrict__ x, const float* __restrict__ w,
    bf16_t* __restrict__ out, int C)
{
  const int row = blockIdx.x;
  const float* xr = x + (long long)row * C;
  __shared__ float red[256];
  float s = 0.f;
  for (int c = threadIdx.x; c < C; c += 256) { float v = xr[c]; s += v*v; }
  red[threadIdx.x] = s; __syncthreads();
  for (int off = 128; off > 0; off >>= 1){
    if (threadIdx.x < off) red[threadIdx.x] += red[threadIdx.x+off];
    __syncthreads();
  }
  const float inv = rsqrtf(red[0] / (float)C + 1e-6f);
  for (int c = threadIdx.x; c < C; c += 256)
    out[(long long)row * C + c] = f2bf(xr[c] * inv * w[c]);
}

// ---------------------------------------------------------------------------
// From projection rows: rope(q,k) -> qbf/kbf [bh][n][32];
// global q/k points (padded to 32) -> qpbf/kpbf [bh][n][32]; kn[bh][n];
// transposed [V | Vpts | 0] B-operand -> vtb [bh][64][512].
// ---------------------------------------------------------------------------
__global__ __launch_bounds__(128) void k_rope_frames(
    const float* __restrict__ proj, const float* __restrict__ rot, const float* __restrict__ trans,
    bf16_t* __restrict__ qbf, bf16_t* __restrict__ kbf,
    bf16_t* __restrict__ qpbf, bf16_t* __restrict__ kpbf,
    bf16_t* __restrict__ vtb, float* __restrict__ kn)
{
  const int bn = blockIdx.x; const int b = bn >> 9; const int n = bn & 511;
  const float* pr = proj + (long long)bn * 1728;
  const float* R  = rot   + bn * 9;
  const float* T  = trans + bn * 3;
  __shared__ float knred[H_*PQ_*3];

  // rope on q and k (half = 16)
  for (int idx = threadIdx.x; idx < H_*16; idx += 128) {
    const int h = idx >> 4, j = idx & 15;
    const float ang = (float)n * expf(-(float)j * (9.210340372f / 16.f));
    const float cs = cosf(ang), sn = sinf(ang);
    long long o = ((long long)(b*H_ + h) * N_ + n) * CH_;
    float x1 = pr[h*CH_ + j], x2 = pr[h*CH_ + j + 16];
    qbf[o + j]      = f2bf(x1*cs - x2*sn);
    qbf[o + j + 16] = f2bf(x1*sn + x2*cs);
    x1 = pr[H_*CH_ + h*CH_ + j]; x2 = pr[H_*CH_ + h*CH_ + j + 16];
    kbf[o + j]      = f2bf(x1*cs - x2*sn);
    kbf[o + j + 16] = f2bf(x1*sn + x2*cs);
  }
  // v -> vtb rows 0..31 (transposed)
  for (int idx = threadIdx.x; idx < H_*CH_; idx += 128) {
    const int h = idx >> 5, c = idx & 31;
    vtb[((long long)(b*H_ + h)*64 + c)*N_ + n] = f2bf(pr[2*H_*CH_ + h*CH_ + c]);
  }
  // q/k points -> global frame, padded to 32
  for (int t = threadIdx.x; t < H_*PQ_*3; t += 128) {
    const int h = t / 12, r = t % 12, p = r / 3, i = r % 3;
    const float* pq = pr + 1152 + h*12 + p*3;
    const float* pk = pr + 1296 + h*12 + p*3;
    float qg = R[i*3+0]*pq[0] + R[i*3+1]*pq[1] + R[i*3+2]*pq[2] + T[i];
    float kg = R[i*3+0]*pk[0] + R[i*3+1]*pk[1] + R[i*3+2]*pk[2] + T[i];
    long long o = ((long long)(b*H_ + h)*N_ + n)*32 + r;
    qpbf[o] = f2bf(qg); kpbf[o] = f2bf(kg);
    knred[t] = kg * kg;
  }
  for (int t = threadIdx.x; t < H_*20; t += 128) {
    const int h = t / 20, c = 12 + t % 20;
    long long o = ((long long)(b*H_ + h)*N_ + n)*32 + c;
    qpbf[o] = f2bf(0.f); kpbf[o] = f2bf(0.f);
  }
  // v points -> global frame -> vtb rows 32..55 ; zero rows 56..63
  for (int t = threadIdx.x; t < H_*PV_*3; t += 128) {
    const int h = t / 24, r = t % 24, p = r / 3, i = r % 3;
    const float* pv = pr + 1440 + h*24 + p*3;
    float vg = R[i*3+0]*pv[0] + R[i*3+1]*pv[1] + R[i*3+2]*pv[2] + T[i];
    vtb[((long long)(b*H_ + h)*64 + 32 + r)*N_ + n] = f2bf(vg);
  }
  for (int t = threadIdx.x; t < H_*8; t += 128) {
    const int h = t / 8, j = t % 8;
    vtb[((long long)(b*H_ + h)*64 + 56 + j)*N_ + n] = f2bf(0.f);
  }
  __syncthreads();
  if (threadIdx.x < H_) {
    float s = 0.f;
#pragma unroll
    for (int j = 0; j < 12; ++j) s += knred[threadIdx.x*12 + j];
    kn[(long long)(b*H_ + threadIdx.x)*N_ + n] = s;
  }
}

// ---------------------------------------------------------------------------
// Add wL * (rms_norm(z) @ w_pair^T) into logits. One block per (b,q); each
// wave handles k rows; zn recomputed on the fly (z read once here).
// ---------------------------------------------------------------------------
__global__ __launch_bounds__(256) void k_bias(
    const float* __restrict__ z, const float* __restrict__ wz,
    const float* __restrict__ wpair, float* __restrict__ L)
{
  const int bq = blockIdx.x; const int b = bq >> 9; const int q = bq & 511;
  __shared__ float swp[H_*CZ_];
  __shared__ float swz[CZ_];
  for (int i = threadIdx.x; i < H_*CZ_; i += 256) swp[i] = wpair[i];
  for (int i = threadIdx.x; i < CZ_;    i += 256) swz[i] = wz[i];
  __syncthreads();
  const int lane = threadIdx.x & 31, wave = threadIdx.x >> 5;
  const float wL = 0.57735026919f;
  for (int k = wave; k < N_; k += 8) {
    const float* zr = z + (((long long)(b*N_ + q))*N_ + k)*CZ_;
    float v[4]; float ss = 0.f;
#pragma unroll
    for (int j = 0; j < 4; ++j) { v[j] = zr[lane + 32*j]; ss += v[j]*v[j]; }
#pragma unroll
    for (int off = 16; off > 0; off >>= 1) ss += __shfl_xor(ss, off, 32);
    const float inv = rsqrtf(ss * (1.f/(float)CZ_) + 1e-6f);
    float acc[H_];
#pragma unroll
    for (int h = 0; h < H_; ++h) acc[h] = 0.f;
#pragma unroll
    for (int j = 0; j < 4; ++j) {
      const int c = lane + 32*j;
      const float zn = v[j] * inv * swz[c];
#pragma unroll
      for (int h = 0; h < H_; ++h) acc[h] += zn * swp[h*CZ_ + c];
    }
#pragma unroll
    for (int h = 0; h < H_; ++h) {
#pragma unroll
      for (int off = 16; off > 0; off >>= 1) acc[h] += __shfl_xor(acc[h], off, 32);
    }
    if (lane == 0) {
#pragma unroll
      for (int h = 0; h < H_; ++h) {
        long long idx = ((long long)(b*H_ + h)*N_ + q)*N_ + k;
        L[idx] += wL * acc[h];
      }
    }
  }
}

// Row softmax over k: logits already hold wL*(att+bias) + wLwCg*dot.
// Subtract 0.5*wLwCg*kn[k] (qn term is a per-row constant -> cancels), mask,
// softmax, emit bf16 probabilities.
__global__ __launch_bounds__(256) void k_softmax(
    const float* __restrict__ L, const float* __restrict__ kn,
    const float* __restrict__ gam, const unsigned char* __restrict__ mask,
    bf16_t* __restrict__ abf)
{
  const int zid = blockIdx.x;           // bh*N + q
  const int bh = zid >> 9, q = zid & 511;
  const int b = bh / H_, h = bh % H_;
  const float ch = 0.5f * gam[h];
  const float* Lr = L + ((long long)bh * N_ + q) * N_;
  const float* knr = kn + (long long)bh * N_;
  const unsigned char* mr = mask + b * N_;
  __shared__ float red[256];
  const int k0 = threadIdx.x, k1 = threadIdx.x + 256;
  float v0 = mr[k0] ? (Lr[k0] - ch * knr[k0]) : -1e9f;
  float v1 = mr[k1] ? (Lr[k1] - ch * knr[k1]) : -1e9f;
  float m = fmaxf(v0, v1);
  red[threadIdx.x] = m; __syncthreads();
  for (int off = 128; off > 0; off >>= 1){
    if (threadIdx.x < off) red[threadIdx.x] = fmaxf(red[threadIdx.x], red[threadIdx.x+off]);
    __syncthreads();
  }
  m = red[0]; __syncthreads();
  float e0 = expf(v0 - m), e1 = expf(v1 - m);
  red[threadIdx.x] = e0 + e1; __syncthreads();
  for (int off = 128; off > 0; off >>= 1){
    if (threadIdx.x < off) red[threadIdx.x] += red[threadIdx.x+off];
    __syncthreads();
  }
  const float rs = 1.f / red[0];
  bf16_t* ar = abf + ((long long)bh * N_ + q) * N_;
  ar[k0] = f2bf(e0 * rs);
  ar[k1] = f2bf(e1 * rs);
}

// opair[b,q,h,c] = sum_k a[b,h,q,k] * zn[b,q,k,c] ; one wave per (b,q);
// zn recomputed (second and last z read). Writes directly into cat cols 768+.
__global__ __launch_bounds__(256) void k_opair(
    const float* __restrict__ z, const float* __restrict__ wz,
    const bf16_t* __restrict__ abf, bf16_t* __restrict__ catbf)
{
  const int gw = blockIdx.x * 8 + (threadIdx.x >> 5); // 0..1023 = b*N+q
  const int lane = threadIdx.x & 31;
  const int b = gw >> 9, q = gw & 511;
  float wzv[4];
#pragma unroll
  for (int j = 0; j < 4; ++j) wzv[j] = wz[lane + 32*j];
  float acc[H_][4];
#pragma unroll
  for (int h = 0; h < H_; ++h)
#pragma unroll
    for (int j = 0; j < 4; ++j) acc[h][j] = 0.f;
  const float* zb = z + ((long long)(b*N_ + q))*N_*CZ_;
  const bf16_t* ab = abf + (((long long)(b*H_))*N_ + q)*N_;
  for (int k = 0; k < N_; ++k) {
    const float* zr = zb + (long long)k * CZ_;
    float v[4]; float ss = 0.f;
#pragma unroll
    for (int j = 0; j < 4; ++j) { v[j] = zr[lane + 32*j]; ss += v[j]*v[j]; }
#pragma unroll
    for (int off = 16; off > 0; off >>= 1) ss += __shfl_xor(ss, off, 32);
    const float inv = rsqrtf(ss * (1.f/(float)CZ_) + 1e-6f);
    float a_h[H_];
#pragma unroll
    for (int h = 0; h < H_; ++h) a_h[h] = bf2f(ab[(long long)h * N_ * N_ + k]);
#pragma unroll
    for (int j = 0; j < 4; ++j) {
      const float zn = v[j] * inv * wzv[j];
#pragma unroll
      for (int h = 0; h < H_; ++h) acc[h][j] += a_h[h] * zn;
    }
  }
  bf16_t* cr = catbf + (long long)gw * 2304 + 768;
#pragma unroll
  for (int h = 0; h < H_; ++h)
#pragma unroll
    for (int j = 0; j < 4; ++j) cr[h*CZ_ + lane + 32*j] = f2bf(acc[h][j]);
}

// Assemble cat cols 0..767: o (384), op_local (288), op_norm (96)
__global__ __launch_bounds__(256) void k_after(
    const float* __restrict__ C64, const float* __restrict__ rot,
    const float* __restrict__ trans, bf16_t* __restrict__ catbf)
{
  const int bn = blockIdx.x; const int b = bn >> 9, n = bn & 511;
  const float* R = rot + bn*9; const float* T = trans + bn*3;
  bf16_t* cr = catbf + (long long)bn * 2304;
  for (int idx = threadIdx.x; idx < H_*CH_; idx += 256) {
    const int h = idx >> 5, c = idx & 31;
    cr[idx] = f2bf(C64[(((long long)(b*H_ + h))*N_ + n)*64 + c]);
  }
  for (int t = threadIdx.x; t < H_*PV_; t += 256) {
    const int h = t >> 3, p = t & 7;
    const float* o = &C64[(((long long)(b*H_ + h))*N_ + n)*64 + 32 + p*3];
    float d0 = o[0] - T[0], d1 = o[1] - T[1], d2 = o[2] - T[2];
    float l0 = R[0]*d0 + R[3]*d1 + R[6]*d2;   // rot^T (op - trans)
    float l1 = R[1]*d0 + R[4]*d1 + R[7]*d2;
    float l2 = R[2]*d0 + R[5]*d1 + R[8]*d2;
    cr[384 + h*24 + p*3 + 0] = f2bf(l0);
    cr[384 + h*24 + p*3 + 1] = f2bf(l1);
    cr[384 + h*24 + p*3 + 2] = f2bf(l2);
    cr[672 + h*8 + p] = f2bf(sqrtf(l0*l0 + l1*l1 + l2*l2 + 1e-8f));
  }
}

__global__ void k_swiglu(const float* __restrict__ h12, bf16_t* __restrict__ g, int n){
  int i = blockIdx.x * blockDim.x + threadIdx.x;
  if (i < n) {
    int r = i / HID_, c = i % HID_;
    float a = h12[(long long)r*2*HID_ + c];
    float b2 = h12[(long long)r*2*HID_ + HID_ + c];
    float si = a / (1.f + expf(-a));
    g[i] = f2bf(si * b2);
  }
}

// Final: rms_norm(s2) @ w_frame^T -> v6 -> frame update; writes all outputs.
__global__ __launch_bounds__(256) void k_final(
    const float* __restrict__ s2, const float* __restrict__ w3n,
    const float* __restrict__ wfr, const float* __restrict__ rot,
    const float* __restrict__ trans,
    float* __restrict__ out_s, float* __restrict__ out_rot, float* __restrict__ out_t)
{
  const int bn = blockIdx.x;
  const float* xr = s2 + (long long)bn * CS_;
  __shared__ float red[256];
  __shared__ float v6[6];
  float ss = 0.f;
  for (int c = threadIdx.x; c < CS_; c += 256){ float v = xr[c]; ss += v*v; }
  red[threadIdx.x] = ss; __syncthreads();
  for (int off = 128; off > 0; off >>= 1){
    if (threadIdx.x < off) red[threadIdx.x] += red[threadIdx.x+off];
    __syncthreads();
  }
  const float inv = rsqrtf(red[0] / (float)CS_ + 1e-6f);
  __syncthreads();
  for (int r = 0; r < 6; ++r) {
    float p = 0.f;
    for (int c = threadIdx.x; c < CS_; c += 256) p += xr[c]*inv*w3n[c]*wfr[r*CS_ + c];
    red[threadIdx.x] = p; __syncthreads();
    for (int off = 128; off > 0; off >>= 1){
      if (threadIdx.x < off) red[threadIdx.x] += red[threadIdx.x+off];
      __syncthreads();
    }
    if (threadIdx.x == 0) v6[r] = red[0];
    __syncthreads();
  }
  for (int c = threadIdx.x; c < CS_; c += 256) out_s[(long long)bn*CS_ + c] = xr[c];
  if (threadIdx.x == 0) {
    float w = 1.f, x = v6[0], y = v6[1], zc = v6[2];
    float qn = rsqrtf(w*w + x*x + y*y + zc*zc);
    w *= qn; x *= qn; y *= qn; zc *= qn;
    float U[9] = {1-2*(y*y+zc*zc), 2*(x*y-w*zc),   2*(x*zc+w*y),
                  2*(x*y+w*zc),   1-2*(x*x+zc*zc), 2*(y*zc-w*x),
                  2*(x*zc-w*y),   2*(y*zc+w*x),   1-2*(x*x+y*y)};
    const float* R = rot + bn*9;
    for (int i = 0; i < 3; ++i)
      for (int j = 0; j < 3; ++j)
        out_rot[bn*9 + i*3 + j] =
          R[i*3+0]*U[0*3+j] + R[i*3+1]*U[1*3+j] + R[i*3+2]*U[2*3+j];
    const float* T = trans + bn*3;
    for (int i = 0; i < 3; ++i)
      out_t[bn*3 + i] = R[i*3+0]*v6[3] + R[i*3+1]*v6[4] + R[i*3+2]*v6[5] + T[i];
  }
}

// ---------------------------------------------------------------------------
extern "C" void kernel_launch(void* const* d_in, const int* in_sizes, int n_in,
                              void* d_out, int out_size, void* d_ws, size_t ws_size,
                              hipStream_t stream) {
  const float* s       = (const float*)d_in[0];
  const float* rot     = (const float*)d_in[1];
  const float* trans   = (const float*)d_in[2];
  const float* z       = (const float*)d_in[3];
  const unsigned char* mask = (const unsigned char*)d_in[4];
  const float* norm_z_w = (const float*)d_in[5];
  const float* norm1_w  = (const float*)d_in[6];
  const float* norm2_w  = (const float*)d_in[7];
  const float* norm3_w  = (const float*)d_in[8];
  const float* w_qkv    = (const float*)d_in[9];
  const float* w_qpts   = (const float*)d_in[10];
  const float* w_kpts   = (const float*)d_in[11];
  const float* w_vpts   = (const float*)d_in[12];
  const float* head_w   = (const float*)d_in[13];
  const float* w_pair   = (const float*)d_in[14];
  const float* w_out    = (const float*)d_in[15];
  const float* w12      = (const float*)d_in[16];
  const float* w3       = (const float*)d_in[17];
  const float* w_frame  = (const float*)d_in[18];

  char* wsb = (char*)d_ws;
  size_t off = 0;
  auto alloc = [&](size_t bytes)->void* {
    void* p = wsb + off;
    off = (off + bytes + 255) & ~(size_t)255;
    return p;
  };
  bf16_t* wcat_bf = (bf16_t*)alloc((size_t)1728*384*sizeof(bf16_t));
  bf16_t* wout_bf = (bf16_t*)alloc((size_t)384*2304*sizeof(bf16_t));
  bf16_t* w12_bf  = (bf16_t*)alloc((size_t)3072*384*sizeof(bf16_t));
  bf16_t* w3_bf   = (bf16_t*)alloc((size_t)384*1536*sizeof(bf16_t));
  bf16_t* s0bf    = (bf16_t*)alloc((size_t)BN_*CS_*sizeof(bf16_t));
  float*  proj    = (float*) alloc((size_t)BN_*1728*sizeof(float));
  bf16_t* qbf     = (bf16_t*)alloc((size_t)BH_*N_*32*sizeof(bf16_t));
  bf16_t* kbf     = (bf16_t*)alloc((size_t)BH_*N_*32*sizeof(bf16_t));
  bf16_t* qpbf    = (bf16_t*)alloc((size_t)BH_*N_*32*sizeof(bf16_t));
  bf16_t* kpbf    = (bf16_t*)alloc((size_t)BH_*N_*32*sizeof(bf16_t));
  bf16_t* vtb     = (bf16_t*)alloc((size_t)BH_*64*N_*sizeof(bf16_t));
  float*  kn      = (float*) alloc((size_t)BH_*N_*sizeof(float));
  float*  gammaA  = (float*) alloc((size_t)BH_*sizeof(float));
  float*  Lbuf    = (float*) alloc((size_t)BH_*N_*N_*sizeof(float));
  bf16_t* abf     = (bf16_t*)alloc((size_t)BH_*N_*N_*sizeof(bf16_t));
  float*  C64     = (float*) alloc((size_t)BH_*N_*64*sizeof(float));
  bf16_t* catbf   = (bf16_t*)alloc((size_t)BN_*2304*sizeof(bf16_t));
  float*  s1      = (float*) alloc((size_t)BN_*CS_*sizeof(float));
  bf16_t* xbf     = (bf16_t*)alloc((size_t)BN_*CS_*sizeof(bf16_t));
  float*  h12     = (float*) alloc((size_t)BN_*2*HID_*sizeof(float));
  bf16_t* gbf     = (bf16_t*)alloc((size_t)BN_*HID_*sizeof(bf16_t));
  float*  s2      = (float*) alloc((size_t)BN_*CS_*sizeof(float));

  auto cvt = [&](const float* src, bf16_t* dst, int n){
    k_f32_to_bf16<<<(n + 255)/256, 256, 0, stream>>>(src, dst, n);
  };
  // weights -> bf16 (w_qkv/qpts/kpts/vpts fused contiguously for one GEMM)
  cvt(w_qkv,  wcat_bf,              1152*384);
  cvt(w_qpts, wcat_bf + 1152*384,   144*384);
  cvt(w_kpts, wcat_bf + 1296*384,   144*384);
  cvt(w_vpts, wcat_bf + 1440*384,   288*384);
  cvt(w_out,  wout_bf, 384*2304);
  cvt(w12,    w12_bf,  3072*384);
  cvt(w3,     w3_bf,   384*1536);

  k_gamma<<<1, 32, 0, stream>>>(head_w, gammaA);
  k_rmsnorm_bf16<<<BN_, 256, 0, stream>>>(s, norm1_w, s0bf, CS_);

  // fused projection GEMM: (1024 x 384) @ (1728 x 384)^T
  k_gemm_bf16<<<dim3(1728/64, BN_/64, 1), 256, 0, stream>>>(
      s0bf, wcat_bf, proj, BN_, 1728, 384, 0, 0, 0, 1.f, nullptr, 0.f);

  k_rope_frames<<<BN_, 128, 0, stream>>>(proj, rot, trans, qbf, kbf, qpbf, kpbf, vtb, kn);

  const float wL = 0.57735026919f;
  // logits = wL/sqrt(CH) * Q K^T  (batched over b*H)
  k_gemm_bf16<<<dim3(8, 8, BH_), 256, 0, stream>>>(
      qbf, kbf, Lbuf, N_, N_, 32,
      (long long)N_*32, (long long)N_*32, (long long)N_*N_,
      wL / sqrtf((float)CH_), nullptr, 0.f);
  // += wL*wC*softplus(head_w[h]) * Qp Kp^T (d2 dot term; qn dropped, kn in softmax)
  k_gemm_bf16<<<dim3(8, 8, BH_), 256, 0, stream>>>(
      qpbf, kpbf, Lbuf, N_, N_, 32,
      (long long)N_*32, (long long)N_*32, (long long)N_*N_,
      1.f, gammaA, 1.f);

  k_bias<<<BN_, 256, 0, stream>>>(z, norm_z_w, w_pair, Lbuf);
  k_softmax<<<BH_*N_, 256, 0, stream>>>(Lbuf, kn, gammaA, mask, abf);

  // [o | op] = A @ [V | Vp | 0]  (batched)
  k_gemm_bf16<<<dim3(1, 8, BH_), 256, 0, stream>>>(
      abf, vtb, C64, N_, 64, N_,
      (long long)N_*N_, (long long)64*N_, (long long)N_*64,
      1.f, nullptr, 0.f);

  k_opair<<<BN_/8, 256, 0, stream>>>(z, norm_z_w, abf, catbf);
  k_after<<<BN_, 256, 0, stream>>>(C64, rot, trans, catbf);

  k_copy_f32<<<(BN_*CS_ + 255)/256, 256, 0, stream>>>(s, s1, BN_*CS_);
  k_gemm_bf16<<<dim3(384/64, BN_/64, 1), 256, 0, stream>>>(
      catbf, wout_bf, s1, BN_, 384, 2304, 0, 0, 0, 1.f, nullptr, 1.f);

  k_rmsnorm_bf16<<<BN_, 256, 0, stream>>>(s1, norm2_w, xbf, CS_);
  k_gemm_bf16<<<dim3(3072/64, BN_/64, 1), 256, 0, stream>>>(
      xbf, w12_bf, h12, BN_, 3072, 384, 0, 0, 0, 1.f, nullptr, 0.f);
  k_swiglu<<<(BN_*HID_ + 255)/256, 256, 0, stream>>>(h12, gbf, BN_*HID_);

  k_copy_f32<<<(BN_*CS_ + 255)/256, 256, 0, stream>>>(s1, s2, BN_*CS_);
  k_gemm_bf16<<<dim3(384/64, BN_/64, 1), 256, 0, stream>>>(
      gbf, w3_bf, s2, BN_, 384, 1536, 0, 0, 0, 1.f, nullptr, 1.f);

  float* out = (float*)d_out;
  k_final<<<BN_, 256, 0, stream>>>(s2, norm3_w, w_frame, rot, trans,
                                   out, out + BN_*CS_, out + BN_*CS_ + BN_*9);
}
